// GAT_39195871543848
// MI455X (gfx1250) — matrix operand order repaired and checked by reading, not compiled, not run on hardware
//
#include <hip/hip_runtime.h>
#include <hip/hip_bf16.h>

// ---------------------------------------------------------------------------
// GAT (3-layer) for MI455X / gfx1250.
//  - GEMMs via v_wmma_f32_16x16x32_f16 (f16 in, f32 accumulate)
//  - B tiles staged into LDS via Tensor Data Mover (tensor_load_to_lds)
//  - Edge softmax/aggregation via L2 atomics (working set < 192MB L2)
// ---------------------------------------------------------------------------

typedef __attribute__((ext_vector_type(16))) _Float16 v16h;
typedef __attribute__((ext_vector_type(8)))  _Float16 v8h;
typedef __attribute__((ext_vector_type(8)))  float    v8f;
typedef __attribute__((ext_vector_type(4)))  unsigned u32x4;
typedef __attribute__((ext_vector_type(8)))  int      i32x8;
typedef __attribute__((ext_vector_type(4)))  int      i32x4;

#define GAT_N      50000
#define GAT_E      800000
#define GAT_E2     850000      /* E + N self-loops */
#define GAT_INC    128
#define GAT_HID    32
#define GAT_HEADS  8
#define GAT_F      256         /* HEADS*HID */
#define GAT_OUTC   40
#define GAT_OUTP   48          /* 40 padded to multiple of 16 */
#define NEG_SLOPE  0.2f
#define ENC_NEG_INF 0x007FFFFFu /* order-preserving encoding of -inf */

// Order-preserving float<->uint mapping so segment-max can use atomicMax(u32).
static __device__ __forceinline__ unsigned enc_ord(float f) {
  unsigned b = __float_as_uint(f);
  return (b & 0x80000000u) ? ~b : (b | 0x80000000u);
}
static __device__ __forceinline__ float dec_ord(unsigned u) {
  return __uint_as_float((u & 0x80000000u) ? (u & 0x7fffffffu) : ~u);
}

// ------------------------------- utility kernels ---------------------------
__global__ void fill_f32(float* p, int n, float v) {
  int i = blockIdx.x * blockDim.x + threadIdx.x;
  if (i < n) p[i] = v;
}
__global__ void fill_u32(unsigned* p, int n, unsigned v) {
  int i = blockIdx.x * blockDim.x + threadIdx.x;
  if (i < n) p[i] = v;
}
__global__ void cvt_f32_f16(const float* __restrict__ s, _Float16* __restrict__ d, int n) {
  int i = blockIdx.x * blockDim.x + threadIdx.x;
  if (i < n) d[i] = (_Float16)s[i];
}
// copy [rows x cols] f32 into a pre-zeroed f16 buffer with row stride ldd
__global__ void cvt_pad_f16(const float* __restrict__ s, _Float16* __restrict__ d,
                            int rows, int cols, int ldd) {
  int i = blockIdx.x * blockDim.x + threadIdx.x;
  if (i >= rows * cols) return;
  int r = i / cols, c = i - r * cols;
  d[r * ldd + c] = (_Float16)s[i];
}

// ------------------------------- WMMA GEMM ---------------------------------
// C[M,Nout] = A[M,K](f16,row-major) * B[K,Nout](f16,row-major), f32 accum.
// Grid: x = row-block (8 waves x 16 rows = 128 rows), y = 16-col tile.
// The block's B[K x 16] tile is staged into LDS by the Tensor Data Mover and
// shared by all 8 waves; A fragments are 16B-vectorized global loads.
__global__ void __launch_bounds__(256)
wmma_gemm_f16(const _Float16* __restrict__ A,
              const _Float16* __restrict__ B,
              float* __restrict__ C,
              int M, int K, int Nout) {
  __shared__ _Float16 ldsB[256 * 16];          // up to K=256 rows of 16 cols
  const int wv     = threadIdx.x >> 5;
  const int lane   = threadIdx.x & 31;
  const int tile_n = blockIdx.y;
  const int tile_m = blockIdx.x * 8 + wv;

  // ---- wave 0 issues one TDM 2-D tile load: B[0:K, tile_n*16 +: 16] -> LDS
  if (wv == 0) {
    unsigned long long ga = (unsigned long long)(B + (size_t)tile_n * 16);
    unsigned ldsa = (unsigned)(size_t)(&ldsB[0]);   // low 32 bits = LDS offset
    u32x4 g0;
    g0[0] = 1u;                                       // count=1 (valid user D#)
    g0[1] = ldsa;                                     // lds_addr
    g0[2] = (unsigned)(ga & 0xffffffffu);             // global_addr[31:0]
    g0[3] = (unsigned)((ga >> 32) & 0x01ffffffu)      // global_addr[56:32]
            | (2u << 30);                             // type=2 ("image")
    i32x8 g1;
    g1[0] = (int)(1u << 16);                          // data_size=1 -> 2 bytes
    g1[1] = (int)(((unsigned)Nout & 0xffffu) << 16);  // tensor_dim0[15:0]
    g1[2] = (int)((((unsigned)Nout >> 16) & 0xffffu)  // tensor_dim0[31:16]
            | (((unsigned)K & 0xffffu) << 16));       // tensor_dim1[15:0]
    g1[3] = (int)((((unsigned)K >> 16) & 0xffffu)     // tensor_dim1[31:16]
            | (16u << 16));                           // tile_dim0 = 16 cols
    g1[4] = K;                                        // tile_dim1=K, tile_dim2=0
    g1[5] = Nout;                                     // tensor_dim0_stride[31:0]
    g1[6] = 0;                                        // stride hi / dim1_stride
    g1[7] = 0;
    i32x4 gz4 = {0, 0, 0, 0};                         // 2-D tensor: groups 2/3 unused
    i32x8 gz8 = {0, 0, 0, 0, 0, 0, 0, 0};
    __builtin_amdgcn_tensor_load_to_lds(g0, g1, gz4, gz4, gz8, 0);
    __builtin_amdgcn_s_wait_tensorcnt(0);
  }
  __syncthreads();

  if (tile_m * 16 < M) {                      // wave-uniform -> EXEC all ones
    const int g = lane >> 4;                  // half-wave group (0/1)
    const int r = lane & 15;
    v8f acc = {};
    const _Float16* Arow = A + (size_t)(tile_m * 16 + r) * K;
    for (int k0 = 0; k0 < K; k0 += 32) {
      if (k0 + 32 < K) __builtin_prefetch(Arow + k0 + 32, 0, 1);
      // ISA 16-bit A 16x32 layout: elems 0-7 -> K=k0+g*8+[0,8),
      //                            elems 8-15 -> K=k0+16+g*8+[0,8): two b128s.
      v8h a0 = *(const v8h*)(Arow + k0 + (g << 3));
      v8h a1 = *(const v8h*)(Arow + k0 + 16 + (g << 3));
      v16h a = __builtin_shufflevector(a0, a1, 0, 1, 2, 3, 4, 5, 6, 7,
                                       8, 9, 10, 11, 12, 13, 14, 15);
      // ISA 16-bit B 32x16 layout: elem i -> K = k0 + g*8 + (i&7) + (i>>3)*16, col = r (LDS)
      v16h b;
#pragma unroll
      for (int i = 0; i < 16; ++i)
        b[i] = ldsB[(k0 + (g << 3) + (i & 7) + ((i >> 3) << 4)) * 16 + r];
      acc = __builtin_amdgcn_wmma_f32_16x16x32_f16(false, a, false, b,
                                                   (short)0, acc, false, false);
    }
    // C layout: VGPR j -> row = g*8 + j, col = r
#pragma unroll
    for (int j = 0; j < 8; ++j)
      C[(size_t)(tile_m * 16 + (g << 3) + j) * Nout + tile_n * 16 + r] = acc[j];
  }
}

// --------------------------- attention kernels -----------------------------
// a_src[n,h] = sum_c h[n,h,c]*att_src[h,c]; same for a_dst.
__global__ void attn_scores_k(const float* __restrict__ h,
                              const float* __restrict__ as,
                              const float* __restrict__ ad,
                              float* __restrict__ a_src,
                              float* __restrict__ a_dst,
                              int n_nodes, int H, int C, int ldh) {
  int idx = blockIdx.x * blockDim.x + threadIdx.x;     // n*H + head
  if (idx >= n_nodes * H) return;
  int n = idx / H, hh = idx - n * H;
  const float* hp  = h + (size_t)n * ldh + hh * C;
  const float* asp = as + hh * C;
  const float* adp = ad + hh * C;
  float s1 = 0.f, s2 = 0.f;
  for (int c = 0; c < C; ++c) { float v = hp[c]; s1 += v * asp[c]; s2 += v * adp[c]; }
  a_src[idx] = s1;
  a_dst[idx] = s2;
}

// e = leaky_relu(a_src[src] + a_dst[dst]); store and segment-max into m[dst].
__global__ void edge_logits_k(const int* __restrict__ ei, int E, int E2, int H,
                              const float* __restrict__ asrc,
                              const float* __restrict__ adst,
                              float* __restrict__ ebuf,
                              unsigned* __restrict__ mmax) {
  int idx = blockIdx.x * blockDim.x + threadIdx.x;     // e*H + head
  if (idx >= E2 * H) return;
  int e = idx / H, hh = idx - e * H;
  int sn, dn;
  if (e < E) { sn = ei[e]; dn = ei[E + e]; } else { sn = dn = e - E; }
  float v = asrc[sn * H + hh] + adst[dn * H + hh];
  v = (v > 0.f) ? v : NEG_SLOPE * v;
  ebuf[idx] = v;
  atomicMax(&mmax[dn * H + hh], enc_ord(v));
}

// ex = exp(e - m[dst]); overwrite ebuf; segment-sum into s[dst].
__global__ void edge_exp_k(const int* __restrict__ ei, int E, int E2, int H,
                           float* __restrict__ ebuf,
                           const unsigned* __restrict__ mmax,
                           float* __restrict__ ssum) {
  int idx = blockIdx.x * blockDim.x + threadIdx.x;
  if (idx >= E2 * H) return;
  int e = idx / H, hh = idx - e * H;
  int dn = (e < E) ? ei[E + e] : (e - E);
  float ex = expf(ebuf[idx] - dec_ord(mmax[dn * H + hh]));
  ebuf[idx] = ex;
  atomicAdd(&ssum[dn * H + hh], ex);
}

// msg = h[src]*alpha, scatter-add into aggr[dst]. One thread per (edge,head).
// C is a multiple of 4 and all row strides keep 16B alignment -> float4 loads.
__global__ void scatter_k(const int* __restrict__ ei, int E, int E2, int H, int C,
                          int ldh, int ldo,
                          const float* __restrict__ ebuf,
                          const float* __restrict__ ssum,
                          const float* __restrict__ hfeat,
                          float* __restrict__ aggr) {
  int idx = blockIdx.x * blockDim.x + threadIdx.x;
  if (idx >= E2 * H) return;
  int e = idx / H, hh = idx - e * H;
  int sn, dn;
  if (e < E) { sn = ei[e]; dn = ei[E + e]; } else { sn = dn = e - E; }
  float alpha = ebuf[idx] / (ssum[dn * H + hh] + 1e-16f);
  const float4* hp4 = (const float4*)(hfeat + (size_t)sn * ldh + hh * C);
  float* op = aggr + (size_t)dn * ldo + hh * C;
  for (int c4 = 0; c4 < (C >> 2); ++c4) {
    float4 hv = hp4[c4];
    atomicAdd(&op[c4 * 4 + 0], hv.x * alpha);
    atomicAdd(&op[c4 * 4 + 1], hv.y * alpha);
    atomicAdd(&op[c4 * 4 + 2], hv.z * alpha);
    atomicAdd(&op[c4 * 4 + 3], hv.w * alpha);
  }
}

// out_f16 = f16(relu(aggr + bias))  (feeds next layer's GEMM)
__global__ void finalize_relu_f16(const float* __restrict__ aggr,
                                  const float* __restrict__ bias,
                                  _Float16* __restrict__ o, int total, int F) {
  int i = blockIdx.x * blockDim.x + threadIdx.x;
  if (i >= total) return;
  float v = aggr[i] + bias[i % F];
  o[i] = (_Float16)(v > 0.f ? v : 0.f);
}

__global__ void finalize_bias_f32(const float* __restrict__ aggr,
                                  const float* __restrict__ bias,
                                  float* __restrict__ o, int total, int F) {
  int i = blockIdx.x * blockDim.x + threadIdx.x;
  if (i >= total) return;
  o[i] = aggr[i] + bias[i % F];
}

__global__ void log_softmax_k(const float* __restrict__ in, float* __restrict__ out,
                              int n_nodes, int C) {
  int n = blockIdx.x * blockDim.x + threadIdx.x;
  if (n >= n_nodes) return;
  const float* p = in + (size_t)n * C;
  float m = -INFINITY;
  for (int c = 0; c < C; ++c) m = fmaxf(m, p[c]);
  float s = 0.f;
  for (int c = 0; c < C; ++c) s += expf(p[c] - m);
  float ls = logf(s);
  float* q = out + (size_t)n * C;
  for (int c = 0; c < C; ++c) q[c] = p[c] - m - ls;
}

// ------------------------------- launcher ----------------------------------
extern "C" void kernel_launch(void* const* d_in, const int* in_sizes, int n_in,
                              void* d_out, int out_size, void* d_ws, size_t ws_size,
                              hipStream_t stream) {
  (void)in_sizes; (void)n_in; (void)out_size; (void)ws_size;
  const float* x   = (const float*)d_in[0];
  const int*   ei  = (const int*)  d_in[1];
  const float* W1  = (const float*)d_in[2];
  const float* as1 = (const float*)d_in[3];
  const float* ad1 = (const float*)d_in[4];
  const float* b1  = (const float*)d_in[5];
  const float* W2  = (const float*)d_in[6];
  const float* as2 = (const float*)d_in[7];
  const float* ad2 = (const float*)d_in[8];
  const float* b2  = (const float*)d_in[9];
  const float* W3  = (const float*)d_in[10];
  const float* as3 = (const float*)d_in[11];
  const float* ad3 = (const float*)d_in[12];
  const float* b3  = (const float*)d_in[13];
  float* out = (float*)d_out;

  // workspace carve-out (~160 MB; fits the 192 MB L2 working set)
  char* w = (char*)d_ws;
  auto take = [&](size_t bytes) { char* p = w; w += (bytes + 255) & ~(size_t)255; return p; };
  _Float16* xf16 = (_Float16*)take((size_t)GAT_N * GAT_F * 2);   // layer input, f16
  _Float16* wf16 = (_Float16*)take((size_t)GAT_F * GAT_F * 2);   // weights, f16
  float*    hbuf = (float*)   take((size_t)GAT_N * GAT_F * 4);   // GEMM output h
  float*    aggr = (float*)   take((size_t)GAT_N * GAT_F * 4);   // aggregated msgs
  float*    asrc = (float*)   take((size_t)GAT_N * GAT_HEADS * 4);
  float*    adst = (float*)   take((size_t)GAT_N * GAT_HEADS * 4);
  unsigned* mmax = (unsigned*)take((size_t)GAT_N * GAT_HEADS * 4);
  float*    ssum = (float*)   take((size_t)GAT_N * GAT_HEADS * 4);
  float*    ebuf = (float*)   take((size_t)GAT_E2 * GAT_HEADS * 4);
  float*    out3 = (float*)   take((size_t)GAT_N * GAT_OUTC * 4);

  auto B = [](long long n) { return dim3((unsigned)((n + 255) / 256)); };
  dim3 T(256);
  const int NH  = GAT_N * GAT_HEADS;
  const int EH  = GAT_E2 * GAT_HEADS;
  const int NF  = GAT_N * GAT_F;
  const int MB  = (GAT_N / 16 + 7) / 8;    // row-blocks of 8 waves (128 rows)

  // ============================ Layer 1 ====================================
  cvt_f32_f16<<<B((long long)GAT_N * GAT_INC), T, 0, stream>>>(x, xf16, GAT_N * GAT_INC);
  cvt_f32_f16<<<B((long long)GAT_INC * GAT_F), T, 0, stream>>>(W1, wf16, GAT_INC * GAT_F);
  wmma_gemm_f16<<<dim3(MB, GAT_F / 16), T, 0, stream>>>(xf16, wf16, hbuf,
                                                        GAT_N, GAT_INC, GAT_F);
  attn_scores_k<<<B(NH), T, 0, stream>>>(hbuf, as1, ad1, asrc, adst,
                                         GAT_N, GAT_HEADS, GAT_HID, GAT_F);
  fill_u32<<<B(NH), T, 0, stream>>>(mmax, NH, ENC_NEG_INF);
  fill_f32<<<B(NH), T, 0, stream>>>(ssum, NH, 0.f);
  fill_f32<<<B(NF), T, 0, stream>>>(aggr, NF, 0.f);
  edge_logits_k<<<B(EH), T, 0, stream>>>(ei, GAT_E, GAT_E2, GAT_HEADS, asrc, adst, ebuf, mmax);
  edge_exp_k<<<B(EH), T, 0, stream>>>(ei, GAT_E, GAT_E2, GAT_HEADS, ebuf, mmax, ssum);
  scatter_k<<<B(EH), T, 0, stream>>>(ei, GAT_E, GAT_E2, GAT_HEADS, GAT_HID,
                                     GAT_F, GAT_F, ebuf, ssum, hbuf, aggr);
  finalize_relu_f16<<<B(NF), T, 0, stream>>>(aggr, b1, xf16, NF, GAT_F);

  // ============================ Layer 2 ====================================
  cvt_f32_f16<<<B((long long)GAT_F * GAT_F), T, 0, stream>>>(W2, wf16, GAT_F * GAT_F);
  wmma_gemm_f16<<<dim3(MB, GAT_F / 16), T, 0, stream>>>(xf16, wf16, hbuf,
                                                        GAT_N, GAT_F, GAT_F);
  attn_scores_k<<<B(NH), T, 0, stream>>>(hbuf, as2, ad2, asrc, adst,
                                         GAT_N, GAT_HEADS, GAT_HID, GAT_F);
  fill_u32<<<B(NH), T, 0, stream>>>(mmax, NH, ENC_NEG_INF);
  fill_f32<<<B(NH), T, 0, stream>>>(ssum, NH, 0.f);
  fill_f32<<<B(NF), T, 0, stream>>>(aggr, NF, 0.f);
  edge_logits_k<<<B(EH), T, 0, stream>>>(ei, GAT_E, GAT_E2, GAT_HEADS, asrc, adst, ebuf, mmax);
  edge_exp_k<<<B(EH), T, 0, stream>>>(ei, GAT_E, GAT_E2, GAT_HEADS, ebuf, mmax, ssum);
  scatter_k<<<B(EH), T, 0, stream>>>(ei, GAT_E, GAT_E2, GAT_HEADS, GAT_HID,
                                     GAT_F, GAT_F, ebuf, ssum, hbuf, aggr);
  finalize_relu_f16<<<B(NF), T, 0, stream>>>(aggr, b2, xf16, NF, GAT_F);

  // ============================ Layer 3 (H=1, C=40, padded to 48) ==========
  fill_u32<<<B(GAT_F * GAT_OUTP / 2), T, 0, stream>>>((unsigned*)wf16,
                                                      GAT_F * GAT_OUTP / 2, 0u);
  cvt_pad_f16<<<B((long long)GAT_F * GAT_OUTC), T, 0, stream>>>(W3, wf16,
                                                                GAT_F, GAT_OUTC, GAT_OUTP);
  wmma_gemm_f16<<<dim3(MB, GAT_OUTP / 16), T, 0, stream>>>(xf16, wf16, hbuf,
                                                           GAT_N, GAT_F, GAT_OUTP);
  attn_scores_k<<<B(GAT_N), T, 0, stream>>>(hbuf, as3, ad3, asrc, adst,
                                            GAT_N, 1, GAT_OUTC, GAT_OUTP);
  fill_u32<<<B(GAT_N), T, 0, stream>>>(mmax, GAT_N, ENC_NEG_INF);
  fill_f32<<<B(GAT_N), T, 0, stream>>>(ssum, GAT_N, 0.f);
  fill_f32<<<B((long long)GAT_N * GAT_OUTC), T, 0, stream>>>(aggr, GAT_N * GAT_OUTC, 0.f);
  edge_logits_k<<<B(GAT_E2), T, 0, stream>>>(ei, GAT_E, GAT_E2, 1, asrc, adst, ebuf, mmax);
  edge_exp_k<<<B(GAT_E2), T, 0, stream>>>(ei, GAT_E, GAT_E2, 1, ebuf, mmax, ssum);
  scatter_k<<<B(GAT_E2), T, 0, stream>>>(ei, GAT_E, GAT_E2, 1, GAT_OUTC,
                                         GAT_OUTP, GAT_OUTC, ebuf, ssum, hbuf, aggr);
  finalize_bias_f32<<<B((long long)GAT_N * GAT_OUTC), T, 0, stream>>>(aggr, b3, out3,
                                                                      GAT_N * GAT_OUTC, GAT_OUTC);
  log_softmax_k<<<B(GAT_N), T, 0, stream>>>(out3, out, GAT_N, GAT_OUTC);
}
